// MultiHeadAttention_35158602285149
// MI455X (gfx1250) — matrix operand order repaired and checked by reading, not compiled, not run on hardware
//
#include <hip/hip_runtime.h>
#include <hip/hip_bf16.h>

// MHA forward for B=2, L=2048, D=1024, H=16, DH=64 on gfx1250 (MI455X).
// All GEMMs use v_wmma_f32_16x16x32_f16 (f16 in, f32 accumulate).
// attn [B,H,L,L] is materialized directly in d_out and softmaxed in place to
// minimize traffic on the 536MB tensor (the HBM-bound part of this problem).

#define Bsz 2
#define Lsz 2048
#define Dsz 1024
#define Hsz 16
#define DHsz 64
#define LDS_S 40   // halves per LDS row (80B: multiple of 16B, staggers banks)

typedef _Float16 h8  __attribute__((ext_vector_type(8)));
typedef _Float16 h16 __attribute__((ext_vector_type(16)));
typedef float    f8  __attribute__((ext_vector_type(8)));

// A-fragment: 16x32 f16, lane = M row (lane&15); half-wave selects K groups.
// VGPR0-3: K = 8*hf .. 8*hf+7 ; VGPR4-7: K = 16+8*hf .. 16+8*hf+7
__device__ __forceinline__ h16 frag_a(const _Float16* __restrict__ A, int lane) {
  const int m = lane & 15, hf = lane >> 4;
  h8 lo = *(const h8*)(A + m * LDS_S + 8 * hf);
  h8 hi = *(const h8*)(A + m * LDS_S + 16 + 8 * hf);
  return __builtin_shufflevector(lo, hi, 0,1,2,3,4,5,6,7,8,9,10,11,12,13,14,15);
}

// B-fragment from K-transposed LDS tile Bt[n][k]: lane = N col (lane&15),
// elements 0-7: K = 8*hf .. 8*hf+7 ; 8-15: K = 16+8*hf .. 16+8*hf+7 (two ds_load_b128).
__device__ __forceinline__ h16 frag_b(const _Float16* __restrict__ Bt, int lane) {
  const int n = lane & 15, hf = lane >> 4;
  h8 lo = *(const h8*)(Bt + n * LDS_S + 8 * hf);
  h8 hi = *(const h8*)(Bt + n * LDS_S + 16 + 8 * hf);
  return __builtin_shufflevector(lo, hi, 0,1,2,3,4,5,6,7,8,9,10,11,12,13,14,15);
}

#define WMMA_F16(a, b, c) \
  __builtin_amdgcn_wmma_f32_16x16x32_f16(false, (a), false, (b), (short)0, (c), false, false)

// ---------------------------------------------------------------------------
// Kernel 1: QKV = x[B*L, D] @ Wqkv[D, 3D] + bqkv ; scatter to Q/K/V f16
// [B,H,L,DH] head-major buffers. Q is pre-scaled by 1/sqrt(DH)=0.125.
// grid: (3D/64, B*L/64), block 128.
// ---------------------------------------------------------------------------
__global__ void __launch_bounds__(128)
k_qkv(const float* __restrict__ x, const float* __restrict__ Wqkv,
      const float* __restrict__ bqkv, _Float16* __restrict__ Qw,
      _Float16* __restrict__ Kw, _Float16* __restrict__ Vw) {
  __shared__ __align__(16) _Float16 Al[64 * LDS_S];
  __shared__ __align__(16) _Float16 Bl[64 * LDS_S];
  const int tid = threadIdx.x, w = tid >> 5, lane = tid & 31;
  const int m0 = blockIdx.y * 64;          // over B*L
  const int n0 = blockIdx.x * 64;          // over 3D
  f8 acc[4] = {};
  for (int k0 = 0; k0 < Dsz; k0 += 32) {
    if (k0 + 32 < Dsz)
      __builtin_prefetch(&x[(m0 + (tid >> 5)) * Dsz + k0 + 32], 0, 1);
    for (int i = tid; i < 64 * 32; i += 128) {          // A: 64 rows x 32 k
      int r = i >> 5, kk = i & 31;
      Al[r * LDS_S + kk] = (_Float16)x[(m0 + r) * Dsz + k0 + kk];
    }
    for (int i = tid; i < 64 * 32; i += 128) {          // B transposed: Bl[n][k]
      int kk = i >> 6, n = i & 63;                      // contiguous n -> coalesced
      Bl[n * LDS_S + kk] = (_Float16)Wqkv[(k0 + kk) * (3 * Dsz) + n0 + n];
    }
    __syncthreads();
    h16 a = frag_a(Al + 16 * w * LDS_S, lane);
#pragma unroll
    for (int nt = 0; nt < 4; ++nt) {
      h16 b = frag_b(Bl + 16 * nt * LDS_S, lane);
      acc[nt] = WMMA_F16(a, b, acc[nt]);
    }
    __syncthreads();
  }
  const int hf = lane >> 4, nl = lane & 15;
#pragma unroll
  for (int nt = 0; nt < 4; ++nt) {
#pragma unroll
    for (int r = 0; r < 8; ++r) {
      int row = m0 + 16 * w + r + 8 * hf;          // b*L + l
      int col = n0 + 16 * nt + nl;                 // 0..3071
      float v = acc[nt][r] + bqkv[col];
      int bb = row >> 11, l = row & (Lsz - 1);
      if (col < Dsz) {
        int h = col >> 6, dh = col & 63;
        Qw[(((size_t)(bb * Hsz + h) * Lsz) + l) * DHsz + dh] = (_Float16)(v * 0.125f);
      } else if (col < 2 * Dsz) {
        int c = col - Dsz, h = c >> 6, dh = c & 63;
        Kw[(((size_t)(bb * Hsz + h) * Lsz) + l) * DHsz + dh] = (_Float16)v;
      } else {
        int c = col - 2 * Dsz, h = c >> 6, dh = c & 63;
        Vw[(((size_t)(bb * Hsz + h) * Lsz) + l) * DHsz + dh] = (_Float16)v;
      }
    }
  }
}

// ---------------------------------------------------------------------------
// Kernel 2: scores[b,h,q,k] = Q·K^T (scale folded into Q), mask applied,
// written straight into the attn region of d_out. grid: (L/64, L/64, B*H).
// ---------------------------------------------------------------------------
__global__ void __launch_bounds__(128)
k_scores(const _Float16* __restrict__ Qw, const _Float16* __restrict__ Kw,
         const int* __restrict__ mask, float* __restrict__ attn) {
  __shared__ __align__(16) _Float16 Al[64 * LDS_S];
  __shared__ __align__(16) _Float16 Bl[64 * LDS_S];
  const int tid = threadIdx.x, w = tid >> 5, lane = tid & 31;
  const int bh = blockIdx.z;
  const int m0 = blockIdx.y * 64, n0 = blockIdx.x * 64;
  const _Float16* Q = Qw + (size_t)bh * Lsz * DHsz;
  const _Float16* K = Kw + (size_t)bh * Lsz * DHsz;
  f8 acc[4] = {};
#pragma unroll
  for (int k0 = 0; k0 < DHsz; k0 += 32) {
    for (int i = tid; i < 64 * 32; i += 128) {
      int r = i >> 5, kk = i & 31;
      Al[r * LDS_S + kk] = Q[(m0 + r) * DHsz + k0 + kk];
    }
    for (int i = tid; i < 64 * 32; i += 128) {       // B = K^T: Bl[n][k]=K[n][k]
      int n = i >> 5, kk = i & 31;
      Bl[n * LDS_S + kk] = K[(n0 + n) * DHsz + k0 + kk];
    }
    __syncthreads();
    h16 a = frag_a(Al + 16 * w * LDS_S, lane);
#pragma unroll
    for (int nt = 0; nt < 4; ++nt) {
      h16 b = frag_b(Bl + 16 * nt * LDS_S, lane);
      acc[nt] = WMMA_F16(a, b, acc[nt]);
    }
    __syncthreads();
  }
  const int hf = lane >> 4, nl = lane & 15;
  const int* M = mask + (size_t)(bh >> 4) * Lsz * Lsz;   // mask[b,0,:,:]
  float* out = attn + (size_t)bh * Lsz * Lsz;
#pragma unroll
  for (int nt = 0; nt < 4; ++nt) {
#pragma unroll
    for (int r = 0; r < 8; ++r) {
      int row = m0 + 16 * w + r + 8 * hf;
      int col = n0 + 16 * nt + nl;
      size_t idx = (size_t)row * Lsz + col;
      float v = acc[nt][r];
      if (M[idx] == 0) v = -1000000000.0f;
      out[idx] = v;
    }
  }
}

// ---------------------------------------------------------------------------
// Kernel 3: row softmax in place on attn (row length L=2048).
// One 256-thread block per row; 8 values/thread.
// ---------------------------------------------------------------------------
__global__ void __launch_bounds__(256)
k_softmax(float* __restrict__ attn) {
  __shared__ float red[256];
  float* p = attn + (size_t)blockIdx.x * Lsz;
  const int t = threadIdx.x;
  float vals[8], mx = -3.4e38f;
#pragma unroll
  for (int i = 0; i < 8; ++i) { vals[i] = p[t + 256 * i]; mx = fmaxf(mx, vals[i]); }
  red[t] = mx; __syncthreads();
  for (int s = 128; s > 0; s >>= 1) {
    if (t < s) red[t] = fmaxf(red[t], red[t + s]);
    __syncthreads();
  }
  mx = red[0]; __syncthreads();
  float sum = 0.0f;
#pragma unroll
  for (int i = 0; i < 8; ++i) { vals[i] = __expf(vals[i] - mx); sum += vals[i]; }
  red[t] = sum; __syncthreads();
  for (int s = 128; s > 0; s >>= 1) {
    if (t < s) red[t] += red[t + s];
    __syncthreads();
  }
  float inv = 1.0f / red[0];
#pragma unroll
  for (int i = 0; i < 8; ++i) p[t + 256 * i] = vals[i] * inv;
}

// ---------------------------------------------------------------------------
// Kernel 4: out = attn[L,L] @ V[L,DH] per (b,h); stored f16 directly in
// concat layout [B, L, H*DH]. grid: (L/64, 1, B*H). N tile = DH = 64.
// ---------------------------------------------------------------------------
__global__ void __launch_bounds__(128)
k_attnv(const float* __restrict__ attn, const _Float16* __restrict__ Vw,
        _Float16* __restrict__ Cc) {
  __shared__ __align__(16) _Float16 Al[64 * LDS_S];
  __shared__ __align__(16) _Float16 Bl[64 * LDS_S];
  const int tid = threadIdx.x, w = tid >> 5, lane = tid & 31;
  const int bh = blockIdx.z;
  const int m0 = blockIdx.x * 64;
  const float* A = attn + (size_t)bh * Lsz * Lsz;
  const _Float16* V = Vw + (size_t)bh * Lsz * DHsz;
  const int bb = bh >> 4, h = bh & 15;
  f8 acc[4] = {};
  for (int k0 = 0; k0 < Lsz; k0 += 32) {
    if (k0 + 32 < Lsz)
      __builtin_prefetch(&A[(size_t)(m0 + (tid >> 5)) * Lsz + k0 + 32], 0, 1);
    for (int i = tid; i < 64 * 32; i += 128) {            // A: attn f32 -> f16
      int r = i >> 5, kk = i & 31;
      Al[r * LDS_S + kk] = (_Float16)A[(size_t)(m0 + r) * Lsz + k0 + kk];
    }
    for (int i = tid; i < 64 * 32; i += 128) {            // B transposed: Bl[n][k]=V[k][n]
      int kk = i >> 6, n = i & 63;                        // contiguous n -> coalesced
      Bl[n * LDS_S + kk] = V[(k0 + kk) * DHsz + n];
    }
    __syncthreads();
    h16 a = frag_a(Al + 16 * w * LDS_S, lane);
#pragma unroll
    for (int nt = 0; nt < 4; ++nt) {
      h16 b = frag_b(Bl + 16 * nt * LDS_S, lane);
      acc[nt] = WMMA_F16(a, b, acc[nt]);
    }
    __syncthreads();
  }
  const int hf = lane >> 4, nl = lane & 15;
#pragma unroll
  for (int nt = 0; nt < 4; ++nt) {
#pragma unroll
    for (int r = 0; r < 8; ++r) {
      int l = m0 + 16 * w + r + 8 * hf;
      int dh = 16 * nt + nl;
      Cc[((size_t)(bb * Lsz + l)) * Dsz + h * DHsz + dh] = (_Float16)acc[nt][r];
    }
  }
}

// ---------------------------------------------------------------------------
// Kernel 5: output = concat[B*L, D] @ Wout[D, D] + bout, f32 to d_out.
// grid: (D/64, B*L/64).
// ---------------------------------------------------------------------------
__global__ void __launch_bounds__(128)
k_outproj(const _Float16* __restrict__ Cc, const float* __restrict__ Wout,
          const float* __restrict__ bout, float* __restrict__ out) {
  __shared__ __align__(16) _Float16 Al[64 * LDS_S];
  __shared__ __align__(16) _Float16 Bl[64 * LDS_S];
  const int tid = threadIdx.x, w = tid >> 5, lane = tid & 31;
  const int m0 = blockIdx.y * 64, n0 = blockIdx.x * 64;
  f8 acc[4] = {};
  for (int k0 = 0; k0 < Dsz; k0 += 32) {
    if (k0 + 32 < Dsz)
      __builtin_prefetch(&Wout[(size_t)(k0 + 32 + (tid >> 2)) * Dsz + n0], 0, 1);
    for (int i = tid; i < 64 * 32; i += 128) {
      int r = i >> 5, kk = i & 31;
      Al[r * LDS_S + kk] = Cc[(size_t)(m0 + r) * Dsz + k0 + kk];
    }
    for (int i = tid; i < 64 * 32; i += 128) {
      int kk = i >> 6, n = i & 63;
      Bl[n * LDS_S + kk] = (_Float16)Wout[(size_t)(k0 + kk) * Dsz + n0 + n];
    }
    __syncthreads();
    h16 a = frag_a(Al + 16 * w * LDS_S, lane);
#pragma unroll
    for (int nt = 0; nt < 4; ++nt) {
      h16 b = frag_b(Bl + 16 * nt * LDS_S, lane);
      acc[nt] = WMMA_F16(a, b, acc[nt]);
    }
    __syncthreads();
  }
  const int hf = lane >> 4, nl = lane & 15;
#pragma unroll
  for (int nt = 0; nt < 4; ++nt) {
#pragma unroll
    for (int r = 0; r < 8; ++r) {
      int row = m0 + 16 * w + r + 8 * hf;
      int col = n0 + 16 * nt + nl;
      out[(size_t)row * Dsz + col] = acc[nt][r] + bout[col];
    }
  }
}

// ---------------------------------------------------------------------------
extern "C" void kernel_launch(void* const* d_in, const int* in_sizes, int n_in,
                              void* d_out, int out_size, void* d_ws, size_t ws_size,
                              hipStream_t stream) {
  const float* x    = (const float*)d_in[0];
  const int*   mask = (const int*)d_in[1];
  const float* Wqkv = (const float*)d_in[2];
  const float* bqkv = (const float*)d_in[3];
  const float* Wout = (const float*)d_in[4];
  const float* bout = (const float*)d_in[5];

  float* out  = (float*)d_out;                       // [B,L,D] = 4,194,304 f32
  float* attn = out + (size_t)Bsz * Lsz * Dsz;       // [B,H,L,L] f32

  const size_t HEADSZ = (size_t)Bsz * Hsz * Lsz * DHsz;  // 4,194,304 f16 each
  _Float16* Qw = (_Float16*)d_ws;
  _Float16* Kw = Qw + HEADSZ;
  _Float16* Vw = Kw + HEADSZ;
  _Float16* Cc = Qw;   // Q dead after k_scores; reuse its 8MB for concat (stream-ordered)

  // 1) QKV projection: M=B*L=4096, N=3D=3072
  k_qkv<<<dim3(3 * Dsz / 64, Bsz * Lsz / 64), 128, 0, stream>>>(x, Wqkv, bqkv, Qw, Kw, Vw);
  // 2) scores + mask -> attn region of d_out: per (b,h) 2048x2048
  k_scores<<<dim3(Lsz / 64, Lsz / 64, Bsz * Hsz), 128, 0, stream>>>(Qw, Kw, mask, attn);
  // 3) softmax in place (B*H*L rows of length L)
  k_softmax<<<dim3(Bsz * Hsz * Lsz), 256, 0, stream>>>(attn);
  // 4) attn @ V -> concat f16 [B, L, H*DH]
  k_attnv<<<dim3(Lsz / 64, 1, Bsz * Hsz), 128, 0, stream>>>(attn, Vw, Cc);
  // 5) output projection + bias -> d_out[0 .. B*L*D)
  k_outproj<<<dim3(Dsz / 64, Bsz * Lsz / 64), 128, 0, stream>>>(Cc, Wout, bout, out);
}
